// ScaledDotProductAttention_49074296324370
// MI455X (gfx1250) — compile-verified, matrix operand order checked by reading, not run on hardware
//
#include <hip/hip_runtime.h>
#include <stdint.h>

// Problem constants (from reference): B=16, Q=2048, S=2048, D=128
#define BB 16
#define QQ 2048
#define SS 2048
#define DD 128
#define QT 32                 // q rows per workgroup
#define SC 64                 // s columns per chunk
#define NCHUNK (SS / SC)      // 32

// padded LDS row strides (ushort elements); multiples of 8 (16B) for uint4 frag loads
#define QS_STRIDE (DD + 8)    // 136 -> 272B rows, bank stride 4
#define KB_STRIDE (DD + 8)    // 136
#define VT_STRIDE (SC + 8)    // 72  -> 144B rows, bank stride 36
#define SB_STRIDE (SC + 8)    // 72

typedef __bf16 v16bf __attribute__((ext_vector_type(16)));
typedef float  v8f   __attribute__((ext_vector_type(8)));

__device__ __forceinline__ unsigned short f32_to_bf16(float f) {
    uint32_t u = __float_as_uint(f);
    uint32_t r = u + 0x7FFFu + ((u >> 16) & 1u);   // round-to-nearest-even
    return (unsigned short)(r >> 16);
}

// 16-element bf16 fragment from two 16-byte aligned LDS runs.
__device__ __forceinline__ v16bf ld_frag(const unsigned short* p0, const unsigned short* p1) {
    union { v16bf v; struct { uint4 a, b; } s; } u;
    u.s.a = *(const uint4*)p0;
    u.s.b = *(const uint4*)p1;
    return u.v;
}

// fetch one 64x128 f32 chunk into registers (8 float4 per thread, coalesced)
__device__ __forceinline__ void fetch8(const float* __restrict__ g, int tid, float4 (&reg)[8]) {
    #pragma unroll
    for (int i = 0; i < 8; ++i) {
        int idx = tid + i * 256;
        reg[i] = *(const float4*)(g + (size_t)(idx >> 5) * DD + (idx & 31) * 4);
    }
}

__device__ __forceinline__ void stage_k(const float4 (&reg)[8], int tid,
                                        unsigned short (*kbuf)[KB_STRIDE]) {
    #pragma unroll
    for (int i = 0; i < 8; ++i) {
        int idx = tid + i * 256;
        int row = idx >> 5, c4 = idx & 31;
        uint2 pk;
        pk.x = (uint32_t)f32_to_bf16(reg[i].x) | ((uint32_t)f32_to_bf16(reg[i].y) << 16);
        pk.y = (uint32_t)f32_to_bf16(reg[i].z) | ((uint32_t)f32_to_bf16(reg[i].w) << 16);
        *(uint2*)&kbuf[row][c4 * 4] = pk;
    }
}

__device__ __forceinline__ void stage_vT(const float4 (&reg)[8], int tid,
                                         unsigned short (*vbuf)[VT_STRIDE]) {
    #pragma unroll
    for (int i = 0; i < 8; ++i) {
        int idx = tid + i * 256;
        int row = idx >> 5, c4 = idx & 31;
        vbuf[c4 * 4 + 0][row] = f32_to_bf16(reg[i].x);  // transpose into LDS
        vbuf[c4 * 4 + 1][row] = f32_to_bf16(reg[i].y);
        vbuf[c4 * 4 + 2][row] = f32_to_bf16(reg[i].z);
        vbuf[c4 * 4 + 3][row] = f32_to_bf16(reg[i].w);
    }
}

__global__ __launch_bounds__(256)
void attn_fwd_kernel(const float* __restrict__ Vg, const float* __restrict__ Kg,
                     const float* __restrict__ Qg, const unsigned char* __restrict__ Mg,
                     float* __restrict__ outg, float* __restrict__ attng)
{
    __shared__ __align__(16) unsigned short qs[QT][QS_STRIDE];      // scaled Q tile, bf16
    __shared__ __align__(16) unsigned short kb[2][SC][KB_STRIDE];   // K chunk ping-pong
    __shared__ __align__(16) unsigned short vbT[2][DD][VT_STRIDE];  // V chunk transposed ping-pong
    __shared__ __align__(16) unsigned short sb[QT][SB_STRIDE];      // P chunk, bf16
    __shared__ float wm[2][4][16];
    __shared__ float wl[2][4][16];

    const int qt    = blockIdx.x;
    const int b     = blockIdx.y;
    const int qbase = qt * QT;
    const int tid   = threadIdx.x;
    const int w     = tid >> 5;
    const int lane  = tid & 31;
    const int g     = lane >> 4;     // 16-lane half
    const int ln    = lane & 15;
    const int mt    = w >> 2;        // 0..1 : q-row 16-tile
    const int nt    = w & 3;         // 0..3 : s-col 16-tile within chunk
    const float scale = 0.08838834764831845f;  // D^-0.5 (folded into Q staging)

    const float* kbase = Kg + (size_t)b * SS * DD;
    const float* vbase = Vg + (size_t)b * SS * DD;
    const size_t mrow  = ((size_t)b * QQ + qbase + mt * 16) * (size_t)SS; // mask & attn layout

    // ---- stage Q tile (32x128 f32 -> bf16 * scale) ----
    {
        const float* qg = Qg + ((size_t)b * QQ + qbase) * DD;
        #pragma unroll
        for (int i = 0; i < 4; ++i) {
            int idx = tid + i * 256;
            int row = idx >> 5, c4 = idx & 31;
            float4 f = *(const float4*)(qg + (size_t)row * DD + c4 * 4);
            uint2 pk;
            pk.x = (uint32_t)f32_to_bf16(f.x * scale) | ((uint32_t)f32_to_bf16(f.y * scale) << 16);
            pk.y = (uint32_t)f32_to_bf16(f.z * scale) | ((uint32_t)f32_to_bf16(f.w * scale) << 16);
            *(uint2*)&qs[row][c4 * 4] = pk;
        }
    }

    // ============ Sweep A: per-lane online row max / sumexp (no shuffles in loop) ============
    float m_run[8], l_run[8];
    uint32_t mbits[8];
    #pragma unroll
    for (int r = 0; r < 8; ++r) { m_run[r] = -INFINITY; l_run[r] = 0.0f; mbits[r] = 0u; }

    float4 kreg[8];
    fetch8(kbase, tid, kreg);   // prologue: chunk 0

    for (int ch = 0; ch < NCHUNK; ++ch) {
        stage_k(kreg, tid, kb[ch & 1]);
        if (ch + 1 < NCHUNK) fetch8(kbase + (size_t)(ch + 1) * SC * DD, tid, kreg);
        __syncthreads();   // kb[ch&1] ready; next chunk's global loads in flight

        const int scol = ch * SC + nt * 16 + ln;
        unsigned char mk[8];
        #pragma unroll
        for (int r = 0; r < 8; ++r)
            mk[r] = __builtin_nontemporal_load(Mg + mrow + (size_t)(r + 8 * g) * SS + scol);

        // score tile: (scaled Q) * K^T, contraction over D=128
        v8f c = {0.f,0.f,0.f,0.f,0.f,0.f,0.f,0.f};
        const unsigned short* arow = &qs[mt * 16 + ln][0];
        const unsigned short* brow = &kb[ch & 1][nt * 16 + ln][0];
        #pragma unroll
        for (int dk = 0; dk < 4; ++dk) {
            v16bf a  = ld_frag(arow + dk * 32 + g * 8, arow + dk * 32 + 16 + g * 8);
            v16bf bf = ld_frag(brow + dk * 32 + g * 8, brow + dk * 32 + 16 + g * 8);
            c = __builtin_amdgcn_wmma_f32_16x16x32_bf16(false, a, false, bf, (short)0, c, false, false);
        }

        #pragma unroll
        for (int r = 0; r < 8; ++r) {
            mbits[r] |= (uint32_t)(mk[r] != 0) << ch;          // stash for sweep B
            float s  = mk[r] ? -3.0e38f : c[r];                // finite "-inf": NaN-free math
            float mn = fmaxf(m_run[r], s);
            l_run[r] = l_run[r] * __expf(m_run[r] - mn) + __expf(s - mn);
            m_run[r] = mn;
        }
    }

    // cross-lane (m,l) combine over the 16 columns this wave owns — once, not per chunk
    #pragma unroll
    for (int r = 0; r < 8; ++r) {
        float m = m_run[r], l = l_run[r];
        #pragma unroll
        for (int off = 1; off < 16; off <<= 1) {
            float om = __shfl_xor(m, off, 32);
            float ol = __shfl_xor(l, off, 32);
            float mn = fmaxf(m, om);
            l = l * __expf(m - mn) + ol * __expf(om - mn);
            m = mn;
        }
        m_run[r] = m; l_run[r] = l;
    }
    if (ln == 0) {
        #pragma unroll
        for (int r = 0; r < 8; ++r) {
            wm[mt][nt][r + 8 * g] = m_run[r];
            wl[mt][nt][r + 8 * g] = l_run[r];
        }
    }
    __syncthreads();
    float fm[8], fli[8];
    #pragma unroll
    for (int r = 0; r < 8; ++r) {
        int rr = r + 8 * g;
        float M0 = wm[mt][0][rr], M1 = wm[mt][1][rr], M2 = wm[mt][2][rr], M3 = wm[mt][3][rr];
        float fmx = fmaxf(fmaxf(M0, M1), fmaxf(M2, M3));
        float L = wl[mt][0][rr] * __expf(M0 - fmx) + wl[mt][1][rr] * __expf(M1 - fmx)
                + wl[mt][2][rr] * __expf(M2 - fmx) + wl[mt][3][rr] * __expf(M3 - fmx);
        fm[r]  = fmx;
        fli[r] = 1.0f / L;
    }

    // ============ Sweep B: normalize, emit attention (NT stores), P·V ============
    v8f oacc0 = {0.f,0.f,0.f,0.f,0.f,0.f,0.f,0.f};
    v8f oacc1 = {0.f,0.f,0.f,0.f,0.f,0.f,0.f,0.f};
    const int m2  = mt;
    const int n2a = nt * 2;
    const int n2b = nt * 2 + 1;

    float4 vreg[8];
    fetch8(kbase, tid, kreg);   // prologue: chunk 0
    fetch8(vbase, tid, vreg);

    for (int ch = 0; ch < NCHUNK; ++ch) {
        stage_k(kreg, tid, kb[ch & 1]);
        stage_vT(vreg, tid, vbT[ch & 1]);
        if (ch + 1 < NCHUNK) {
            fetch8(kbase + (size_t)(ch + 1) * SC * DD, tid, kreg);
            fetch8(vbase + (size_t)(ch + 1) * SC * DD, tid, vreg);
        }
        __syncthreads();   // B1: kb/vbT[ch&1] ready

        // recompute score tile (cheaper than round-tripping raw scores through HBM)
        v8f c = {0.f,0.f,0.f,0.f,0.f,0.f,0.f,0.f};
        const unsigned short* arow = &qs[mt * 16 + ln][0];
        const unsigned short* brow = &kb[ch & 1][nt * 16 + ln][0];
        #pragma unroll
        for (int dk = 0; dk < 4; ++dk) {
            v16bf a  = ld_frag(arow + dk * 32 + g * 8, arow + dk * 32 + 16 + g * 8);
            v16bf bf = ld_frag(brow + dk * 32 + g * 8, brow + dk * 32 + 16 + g * 8);
            c = __builtin_amdgcn_wmma_f32_16x16x32_bf16(false, a, false, bf, (short)0, c, false, false);
        }

        const int scol = ch * SC + nt * 16 + ln;
        #pragma unroll
        for (int r = 0; r < 8; ++r) {
            float s = ((mbits[r] >> ch) & 1u) ? -3.0e38f : c[r];   // mask from register stash
            float p = __expf(s - fm[r]) * fli[r];
            __builtin_nontemporal_store(p, attng + mrow + (size_t)(r + 8 * g) * SS + scol);
            sb[mt * 16 + r + 8 * g][nt * 16 + ln] = f32_to_bf16(p);
        }
        __syncthreads();   // B2: sb ready

        // out accumulation: (32x64 P chunk) x (64x128 V); 2 tiles per wave
        const unsigned short* prow = &sb[m2 * 16 + ln][0];
        const unsigned short* v0   = &vbT[ch & 1][n2a * 16 + ln][0];
        const unsigned short* v1   = &vbT[ch & 1][n2b * 16 + ln][0];
        #pragma unroll
        for (int kk = 0; kk < 2; ++kk) {
            v16bf a  = ld_frag(prow + kk * 32 + g * 8, prow + kk * 32 + 16 + g * 8);
            v16bf b0 = ld_frag(v0   + kk * 32 + g * 8, v0   + kk * 32 + 16 + g * 8);
            v16bf b1 = ld_frag(v1   + kk * 32 + g * 8, v1   + kk * 32 + 16 + g * 8);
            oacc0 = __builtin_amdgcn_wmma_f32_16x16x32_bf16(false, a, false, b0, (short)0, oacc0, false, false);
            oacc1 = __builtin_amdgcn_wmma_f32_16x16x32_bf16(false, a, false, b1, (short)0, oacc1, false, false);
        }
    }

    // ---- write out tiles ----
    const size_t obase = ((size_t)b * QQ + qbase + m2 * 16) * (size_t)DD;
    #pragma unroll
    for (int r = 0; r < 8; ++r) {
        outg[obase + (size_t)(r + 8 * g) * DD + n2a * 16 + ln] = oacc0[r];
        outg[obase + (size_t)(r + 8 * g) * DD + n2b * 16 + ln] = oacc1[r];
    }
}

extern "C" void kernel_launch(void* const* d_in, const int* in_sizes, int n_in,
                              void* d_out, int out_size, void* d_ws, size_t ws_size,
                              hipStream_t stream) {
    const float*         Vg = (const float*)d_in[0];          // value [B,S,D]
    const float*         Kg = (const float*)d_in[1];          // key   [B,S,D]
    const float*         Qg = (const float*)d_in[2];          // query [B,Q,D]
    const unsigned char* Mg = (const unsigned char*)d_in[3];  // mask  [B,Q,S] bool
    float* outg  = (float*)d_out;                             // out   [B,Q,D]
    float* attng = outg + (size_t)BB * QQ * DD;               // attention [B,Q,S]

    dim3 grid(QQ / QT, BB);
    attn_fwd_kernel<<<grid, 256, 0, stream>>>(Vg, Kg, Qg, Mg, outg, attng);
}
